// Discriminator_7129645712019
// MI455X (gfx1250) — matrix operand order repaired and checked by reading, not compiled, not run on hardware
//
#include <hip/hip_runtime.h>

// ---------------------------------------------------------------------------
// GRU stack for MI455X (gfx1250): bf16 WMMA GEMMs + persistent recurrent
// kernel (register- or LDS-resident recurrent weights, device-scope barrier).
// ---------------------------------------------------------------------------

typedef __bf16 bf16_t;
typedef __attribute__((ext_vector_type(8)))  bf16_t v8bf;
typedef __attribute__((ext_vector_type(16))) bf16_t v16bf;
typedef __attribute__((ext_vector_type(8)))  float  v8f;

#define WMMA_BF16(a, b, c) \
  __builtin_amdgcn_wmma_f32_16x16x32_bf16(false, (a), false, (b), (short)0, (c), false, false)

__device__ __forceinline__ bf16_t f2bf(float f) {
  unsigned u = __float_as_uint(f);
  u += 0x7FFFu + ((u >> 16) & 1u);           // round to nearest even
  return __builtin_bit_cast(bf16_t, (unsigned short)(u >> 16));
}

__device__ __forceinline__ float sigmoidf_fast(float x) {
  return 1.0f / (1.0f + __expf(-x));
}

// A fragment: 16(M) x 32(K) bf16 from row-major [rows, ld].
// Lane m = L&15; halves split K at 16-granularity (matches load_fragB).
__device__ __forceinline__ v16bf load_fragA(const bf16_t* base, int ld, int k0, int lane) {
  int m = lane & 15, half = lane >> 4;
  const bf16_t* p = base + (size_t)m * ld + k0 + half * 16;
  v8bf lo = *(const v8bf*)p;          // K = k0+half*16 .. +7     (16B)
  v8bf hi = *(const v8bf*)(p + 8);    // K = k0+half*16+8 .. +7   (16B)
  v16bf r;
#pragma unroll
  for (int i = 0; i < 8; ++i) { r[i] = lo[i]; r[i + 8] = hi[i]; }
  return r;
}

// B fragment: 32(K) x 16(N) bf16 read from row-major B^T [N, ld=K].
// Lane n = L&15; halves split K at 16-granularity -> one contiguous 32B load.
__device__ __forceinline__ v16bf load_fragB(const bf16_t* baseT, int ld, int k0, int lane) {
  int n = lane & 15, half = lane >> 4;
  return *(const v16bf*)(baseT + (size_t)n * ld + k0 + half * 16);
}

// ---------------------------------------------------------------------------
// Prep kernels
// ---------------------------------------------------------------------------
__global__ void k_f32_to_bf16(const float* __restrict__ in, bf16_t* __restrict__ out, size_t n) {
  size_t i = (size_t)blockIdx.x * blockDim.x + threadIdx.x;
  if (i < n) out[i] = f2bf(in[i]);
}

// in: [K, N] row-major fp32  ->  out: [N, K] row-major bf16 (transpose+convert)
__global__ void k_transpose_to_bf16(const float* __restrict__ in, bf16_t* __restrict__ out,
                                    int K, int N) {
  size_t i = (size_t)blockIdx.x * blockDim.x + threadIdx.x;
  if (i < (size_t)K * N) {
    int n = (int)(i / K), k = (int)(i % K);
    out[i] = f2bf(in[(size_t)k * N + n]);
  }
}

__global__ void k_zero_u32(unsigned* p, int n) {
  int i = blockIdx.x * blockDim.x + threadIdx.x;
  if (i < n) p[i] = 0u;
}

// ---------------------------------------------------------------------------
// Input projection: C[M,N] = A[M,K](bf16) @ Bt[N,K]^T(bf16) + bias[N], fp32 out
// One wave per 16x64 output strip: A fragment reused across 4 WMMAs per K-step.
// ---------------------------------------------------------------------------
__global__ void k_xproj_wmma(const bf16_t* __restrict__ A,
                             const bf16_t* __restrict__ Bt,
                             const float*  __restrict__ bias,
                             float* __restrict__ C,
                             int M, int N, int K) {
  int wave = blockIdx.x * (blockDim.x >> 5) + (threadIdx.x >> 5);
  int lane = threadIdx.x & 31;
  int tnCnt = N >> 6;                        // 64-wide column supertiles
  int tiles = (M >> 4) * tnCnt;
  if (wave >= tiles) return;                 // wave-uniform: EXEC stays all-1s
  int tm = wave / tnCnt, tn = wave % tnCnt;
  const bf16_t* Abase = A  + (size_t)(tm * 16) * K;
  const bf16_t* Bbase = Bt + (size_t)(tn * 64) * K;

  v8f acc[4] = {{}, {}, {}, {}};
  for (int k0 = 0; k0 < K; k0 += 32) {
    v16bf a = load_fragA(Abase, K, k0, lane);
#pragma unroll
    for (int j = 0; j < 4; ++j) {
      v16bf b = load_fragB(Bbase + (size_t)(j * 16) * K, K, k0, lane);
      acc[j] = WMMA_BF16(a, b, acc[j]);
    }
  }
  int half = lane >> 4, nn = lane & 15;
#pragma unroll
  for (int j = 0; j < 4; ++j) {
    int n = tn * 64 + j * 16 + nn;
    float bv = bias[n];
#pragma unroll
    for (int r = 0; r < 8; ++r) {
      int m = tm * 16 + r + 8 * half;
      C[(size_t)m * N + n] = acc[j][r] + bv;
    }
  }
}

// ---------------------------------------------------------------------------
// Device-scope step barrier for the persistent recurrent kernel.
// ---------------------------------------------------------------------------
__device__ __forceinline__ void grid_sync(unsigned* cnt, unsigned target) {
  __threadfence();
  __syncthreads();
  if (threadIdx.x == 0) {
    __hip_atomic_fetch_add(cnt, 1u, __ATOMIC_ACQ_REL, __HIP_MEMORY_SCOPE_AGENT);
    while (__hip_atomic_load(cnt, __ATOMIC_ACQUIRE, __HIP_MEMORY_SCOPE_AGENT) < target) {
      __builtin_amdgcn_s_sleep(1);
    }
  }
  __syncthreads();
  __threadfence();
}

// ---------------------------------------------------------------------------
// Persistent GRU recurrence (Keras reset_after=True), templated on H.
//   Ut  : [3H, H] bf16 (U transposed)   br : recurrent bias [3H]
//   xg  : [B*T, 3H] fp32 precomputed input projections (+input bias)
//   h   : [B, H] fp32 state; hb0/hb1: bf16 double-buffered shadow of h
//   seq : [B*T, H] bf16 output sequence (nullptr for last layer)
// Grid: H/16 workgroups x 256 threads. WG bi owns h columns [16*bi,16*bi+16);
// its 8 waves own the 8 row-tiles of B=128.
// H<=128: U fragments live in VGPRs for the whole sequence (occupancy is
//         irrelevant here -> registers are free; shortest critical path).
// H==256: U slice staged once into LDS (register copy would need ~192 VGPRs
//         for B-fragments alone and risk spills).
// ACT: 0 = sigmoid, 1 = relu (candidate activation).
// ---------------------------------------------------------------------------
template <int H, int ACT>
__global__ void k_gru_rec(const bf16_t* __restrict__ Ut,
                          const float*  __restrict__ br,
                          const float*  __restrict__ xg,
                          float*  __restrict__ h,
                          bf16_t* __restrict__ hb0,
                          bf16_t* __restrict__ hb1,
                          bf16_t* __restrict__ seq,
                          unsigned* __restrict__ cnt,
                          int T) {
  constexpr int KITER = H / 32;
  constexpr int N3 = 3 * H;
  constexpr bool REGU = (H <= 128);
  __shared__ __align__(32) bf16_t sU[REGU ? 16 : 3 * 16 * H];

  int wid  = threadIdx.x >> 5;
  int lane = threadIdx.x & 31;
  int m0 = wid * 16;                 // row tile (batch)
  int c0 = blockIdx.x * 16;          // column tile (hidden units)
  int half = lane >> 4, nn = lane & 15;
  int col = c0 + nn;
  unsigned nwg = gridDim.x;

  v16bf uz[KITER], ur[KITER], uh[KITER];
  if constexpr (REGU) {
    // Keep this WG's U slice (3 gates x 16 cols x H) in registers.
#pragma unroll
    for (int k = 0; k < KITER; ++k) {
      uz[k] = load_fragB(Ut + (size_t)(0 * H + c0) * H, H, k * 32, lane);
      ur[k] = load_fragB(Ut + (size_t)(1 * H + c0) * H, H, k * 32, lane);
      uh[k] = load_fragB(Ut + (size_t)(2 * H + c0) * H, H, k * 32, lane);
    }
  } else {
    // Stage Ut[{z,r,h} gates, cols c0..c0+15, all K] into LDS (16B chunks).
    constexpr int chunksPerRow = H / 8;
    for (int ci = threadIdx.x; ci < 48 * chunksPerRow; ci += blockDim.x) {
      int r  = ci / chunksPerRow;
      int kc = (ci - r * chunksPerRow) << 3;
      int g = r >> 4, cc = r & 15;
      *(uint4*)&sU[(size_t)r * H + kc] =
          *(const uint4*)&Ut[(size_t)(g * H + c0 + cc) * H + kc];
    }
  }

  // h0 = 0 (each element has exactly one owner lane-slot).
#pragma unroll
  for (int r = 0; r < 8; ++r) {
    int m = m0 + r + 8 * half;
    h[(size_t)m * H + col] = 0.0f;
    hb0[(size_t)m * H + col] = f2bf(0.0f);
  }

  unsigned gen = 1;
  grid_sync(cnt, gen * nwg);

  float bz = br[col], brg = br[H + col], bh = br[2 * H + col];

  for (int t = 0; t < T; ++t) {
    const bf16_t* rd = (t & 1) ? hb1 : hb0;   // read h(t)
    bf16_t*       wr = (t & 1) ? hb0 : hb1;   // write h(t+1)

    v8f cz = {}, cr = {}, ch = {};
#pragma unroll
    for (int k = 0; k < KITER; ++k) {
      v16bf a = load_fragA(rd + (size_t)m0 * H, H, k * 32, lane);
      if constexpr (REGU) {
        cz = WMMA_BF16(a, uz[k], cz);
        cr = WMMA_BF16(a, ur[k], cr);
        ch = WMMA_BF16(a, uh[k], ch);
      } else {
        v16bf bZ = load_fragB(sU,              H, k * 32, lane);
        v16bf bR = load_fragB(sU + 16 * H,     H, k * 32, lane);
        v16bf bH = load_fragB(sU + 32 * H,     H, k * 32, lane);
        cz = WMMA_BF16(a, bZ, cz);
        cr = WMMA_BF16(a, bR, cr);
        ch = WMMA_BF16(a, bH, ch);
      }
    }

#pragma unroll
    for (int r = 0; r < 8; ++r) {
      int m = m0 + r + 8 * half;
      const float* xrow = xg + ((size_t)m * T + t) * N3;
      float z    = sigmoidf_fast(xrow[col]         + cz[r] + bz);
      float rg   = sigmoidf_fast(xrow[H + col]     + cr[r] + brg);
      float cpre = xrow[2 * H + col] + rg * (ch[r] + bh);
      float cand = ACT ? fmaxf(cpre, 0.0f) : sigmoidf_fast(cpre);
      float hold = h[(size_t)m * H + col];
      float hn = z * hold + (1.0f - z) * cand;
      h[(size_t)m * H + col] = hn;
      bf16_t hnb = f2bf(hn);
      wr[(size_t)m * H + col] = hnb;
      if (seq) seq[((size_t)m * T + t) * H + col] = hnb;
    }

    if (t + 1 < T) {  // warm L2/L0 for next step's gate pre-activations
      const float* pf = xg + ((size_t)(m0 + nn) * T + (t + 1)) * N3 + c0;
      __builtin_prefetch(pf, 0, 1);
    }

    ++gen;
    grid_sync(cnt, gen * nwg);
  }
}

// ---------------------------------------------------------------------------
// Head: out[b] = sigmoid(h3[b,:] . Wo + bo)
// ---------------------------------------------------------------------------
__global__ void k_final_dense(const float* __restrict__ h, const float* __restrict__ Wo,
                              const float* __restrict__ bo, float* __restrict__ out, int H) {
  int b = threadIdx.x;
  float acc = bo[0];
  for (int k = 0; k < H; ++k) acc += h[(size_t)b * H + k] * Wo[k];
  out[b] = 1.0f / (1.0f + __expf(-acc));
}

// ---------------------------------------------------------------------------
extern "C" void kernel_launch(void* const* d_in, const int* in_sizes, int n_in,
                              void* d_out, int out_size, void* d_ws, size_t ws_size,
                              hipStream_t stream) {
  const float* text = (const float*)d_in[0];
  const float* W1 = (const float*)d_in[1];
  const float* U1 = (const float*)d_in[2];
  const float* b1 = (const float*)d_in[3];
  const float* W2 = (const float*)d_in[4];
  const float* U2 = (const float*)d_in[5];
  const float* b2 = (const float*)d_in[6];
  const float* W3 = (const float*)d_in[7];
  const float* U3 = (const float*)d_in[8];
  const float* b3 = (const float*)d_in[9];
  const float* Wo = (const float*)d_in[10];
  const float* bo = (const float*)d_in[11];
  float* out = (float*)d_out;

  const int B = 128, T = 512, F = 128;
  const int H1 = 256, H2 = 128, H3 = 64;
  const int M = B * T;

  // Workspace carving (256B aligned).
  char* base = (char*)d_ws;
  size_t off = 0;
  auto carve = [&](size_t bytes) -> char* {
    off = (off + 255) & ~(size_t)255;
    char* p = base + off;
    off += bytes;
    return p;
  };
  bf16_t* Xbf = (bf16_t*)carve((size_t)M * F * 2);        // text in bf16
  bf16_t* S1  = (bf16_t*)carve((size_t)M * H1 * 2);       // layer1 sequence
  bf16_t* S2  = (bf16_t*)carve((size_t)M * H2 * 2);       // layer2 sequence
  float*  XG  = (float*) carve((size_t)M * 3 * H1 * 4);   // xg scratch (reused)
  bf16_t* Wt1 = (bf16_t*)carve((size_t)3 * H1 * F  * 2);
  bf16_t* Ut1 = (bf16_t*)carve((size_t)3 * H1 * H1 * 2);
  bf16_t* Wt2 = (bf16_t*)carve((size_t)3 * H2 * H1 * 2);
  bf16_t* Ut2 = (bf16_t*)carve((size_t)3 * H2 * H2 * 2);
  bf16_t* Wt3 = (bf16_t*)carve((size_t)3 * H3 * H2 * 2);
  bf16_t* Ut3 = (bf16_t*)carve((size_t)3 * H3 * H3 * 2);
  float*  hSt = (float*) carve((size_t)B * H1 * 4);       // fp32 hidden state
  bf16_t* hb0 = (bf16_t*)carve((size_t)B * H1 * 2);
  bf16_t* hb1 = (bf16_t*)carve((size_t)B * H1 * 2);
  unsigned* cnt = (unsigned*)carve(3 * sizeof(unsigned));

  auto cdiv = [](size_t a, size_t b) -> unsigned { return (unsigned)((a + b - 1) / b); };

  // --- prep: barrier counters, bf16 input, transposed bf16 weights ---
  k_zero_u32<<<1, 32, 0, stream>>>(cnt, 3);
  k_f32_to_bf16<<<cdiv((size_t)M * F, 256), 256, 0, stream>>>(text, Xbf, (size_t)M * F);
  k_transpose_to_bf16<<<cdiv((size_t)F  * 3 * H1, 256), 256, 0, stream>>>(W1, Wt1, F,  3 * H1);
  k_transpose_to_bf16<<<cdiv((size_t)H1 * 3 * H1, 256), 256, 0, stream>>>(U1, Ut1, H1, 3 * H1);
  k_transpose_to_bf16<<<cdiv((size_t)H1 * 3 * H2, 256), 256, 0, stream>>>(W2, Wt2, H1, 3 * H2);
  k_transpose_to_bf16<<<cdiv((size_t)H2 * 3 * H2, 256), 256, 0, stream>>>(U2, Ut2, H2, 3 * H2);
  k_transpose_to_bf16<<<cdiv((size_t)H2 * 3 * H3, 256), 256, 0, stream>>>(W3, Wt3, H2, 3 * H3);
  k_transpose_to_bf16<<<cdiv((size_t)H3 * 3 * H3, 256), 256, 0, stream>>>(U3, Ut3, H3, 3 * H3);

  // --- layer 1: xg = text@W1 + b1[0]; recurrence (sigmoid candidate) ---
  {
    size_t tiles = (size_t)(M / 16) * ((3 * H1) / 64);
    k_xproj_wmma<<<cdiv(tiles * 32, 256), 256, 0, stream>>>(Xbf, Wt1, b1, XG, M, 3 * H1, F);
    k_gru_rec<H1, 0><<<H1 / 16, 256, 0, stream>>>(Ut1, b1 + 3 * H1, XG, hSt, hb0, hb1, S1,
                                                  cnt + 0, T);
  }
  // --- layer 2 ---
  {
    size_t tiles = (size_t)(M / 16) * ((3 * H2) / 64);
    k_xproj_wmma<<<cdiv(tiles * 32, 256), 256, 0, stream>>>(S1, Wt2, b2, XG, M, 3 * H2, H1);
    k_gru_rec<H2, 0><<<H2 / 16, 256, 0, stream>>>(Ut2, b2 + 3 * H2, XG, hSt, hb0, hb1, S2,
                                                  cnt + 1, T);
  }
  // --- layer 3 (relu candidate, keep only final h) ---
  {
    size_t tiles = (size_t)(M / 16) * ((3 * H3) / 64);
    k_xproj_wmma<<<cdiv(tiles * 32, 256), 256, 0, stream>>>(S2, Wt3, b3, XG, M, 3 * H3, H2);
    k_gru_rec<H3, 1><<<H3 / 16, 256, 0, stream>>>(Ut3, b3 + 3 * H3, XG, hSt, hb0, hb1, nullptr,
                                                  cnt + 2, T);
  }
  // --- head ---
  k_final_dense<<<1, 128, 0, stream>>>(hSt, Wo, bo, out, H3);
}
